// TMSA_58617713656001
// MI455X (gfx1250) — matrix operand order repaired and checked by reading, not compiled, not run on hardware
//
#include <hip/hip_runtime.h>
#include <hip/hip_bf16.h>

// ---------------------------------------------------------------------------
// TMSA block for MI455X (gfx1250, wave32, WMMA f32_16x16x32_f16)
// B=1, D=6, H=128, W=128, C=180, NH=6, HD=30, WS=(2,8,8), N=128, HID=360
// Packed-fragment weights: [ktile][ntile][lane][16 halves] -> contiguous b128
// fragment loads.  GEMM K-loops are software-pipelined (one K-step lookahead)
// so WMMA overlaps with vmem instead of draining loadcnt to 0 each step.
// ---------------------------------------------------------------------------

typedef __attribute__((ext_vector_type(16))) _Float16 v16h;
typedef __attribute__((ext_vector_type(8)))  float    v8f;

#define NWIN   768        // (6/2)*(128/8)*(128/8)
#define NTOK   128        // tokens per window
#define MTOT   98304      // NWIN*NTOK
#define CDIM   180
#define CPAD   192
#define CKT    6          // CPAD/32 k-tiles
#define NHEAD  6
#define HD     30
#define HDP    32
#define NQKV   540
#define QKV_NT 36         // N tiles (576 cols padded)
#define HID    360
#define MLP_NT 24         // N tiles (384 cols padded)
#define HID2P  384        // K-dim pad of 360
#define H2KT   12         // HID2P/32
#define PRJ_NT 12         // 192 cols padded (N=180)
#define SCALE_F 0.18257418583505536f   // 30^-0.5

static __device__ __forceinline__ v8f wmma16(v16h a, v16h b, v8f c) {
  // v_wmma_f32_16x16x32_f16: (neg_a, A, neg_b, B, c_mod, C, reuse_a, reuse_b)
  return __builtin_amdgcn_wmma_f32_16x16x32_f16(false, a, false, b, (short)0, c,
                                                false, false);
}

// A fragment: 16x32 f16, row-major A[M][K].  Lane layout (this code's contiguous k order, matched by B):
// lanes 0-15 M=0..15 K=0..15; lanes 16-31 same M, K=16..31.
static __device__ __forceinline__ v16h load_A_frag(const _Float16* A, int lda,
                                                   int m0, int k0) {
  int lane = threadIdx.x & 31;
  const _Float16* p = A + (size_t)(m0 + (lane & 15)) * lda + k0 + ((lane >> 4) * 16);
  v16h a;
#pragma unroll
  for (int j = 0; j < 8; ++j) a[j] = p[j];
#pragma unroll
  for (int j = 0; j < 8; ++j) a[8 + j] = p[8 + j];
  return a;
}

// A fragment from f32 source (softmax P in LDS), converted on load.
static __device__ __forceinline__ v16h load_A_frag_f32(const float* A, int lda,
                                                       int m0, int k0) {
  int lane = threadIdx.x & 31;
  const float* p = A + (size_t)(m0 + (lane & 15)) * lda + k0 + ((lane >> 4) * 16);
  v16h a;
#pragma unroll
  for (int j = 0; j < 8; ++j) a[j] = (_Float16)p[j];
#pragma unroll
  for (int j = 0; j < 8; ++j) a[8 + j] = (_Float16)p[8 + j];
  return a;
}

// Packed B fragment: blocks of [32 lanes][16 halves]; per-lane contiguous 32B.
static __device__ __forceinline__ v16h load_B_packed(const _Float16* WP,
                                                     int Ntiles, int kt, int nt) {
  int lane = threadIdx.x & 31;
  const _Float16* p = WP + (((size_t)kt * Ntiles + nt) * 32 + lane) * 16;
  v16h b;
#pragma unroll
  for (int j = 0; j < 16; ++j) b[j] = p[j];
  return b;
}

// B fragment for S = Q*K^T: logical B[k=hd][n=token] = Kmat[token][hd], Kmat
// row-major [tok][HDP]; per-lane 16 contiguous halves.
static __device__ __forceinline__ v16h load_BT_frag(const _Float16* Kmat, int ldk,
                                                    int n0) {
  int lane = threadIdx.x & 31;
  const _Float16* p = Kmat + (size_t)(n0 + (lane & 15)) * ldk + ((lane >> 4) * 16);
  v16h b;
#pragma unroll
  for (int j = 0; j < 16; ++j) b[j] = p[j];
  return b;
}

// Software-pipelined 16x64 GEMM strip: 4 N-tiles, one K-step lookahead.
template <int KT, int NTILES>
static __device__ __forceinline__ void gemm_strip(const _Float16* A, int lda,
                                                  int m0, const _Float16* WP,
                                                  int ntbase, v8f acc[4]) {
  v16h a_cur = load_A_frag(A, lda, m0, 0);
  v16h b_cur[4];
#pragma unroll
  for (int t = 0; t < 4; ++t) b_cur[t] = load_B_packed(WP, NTILES, 0, ntbase + t);
#pragma unroll
  for (int kt = 0; kt < KT; ++kt) {
    int ktn = (kt + 1 < KT) ? kt + 1 : kt;   // clamp: last prefetch is redundant
    v16h a_nxt = load_A_frag(A, lda, m0, ktn * 32);
    v16h b_nxt[4];
#pragma unroll
    for (int t = 0; t < 4; ++t) b_nxt[t] = load_B_packed(WP, NTILES, ktn, ntbase + t);
#pragma unroll
    for (int t = 0; t < 4; ++t) acc[t] = wmma16(a_cur, b_cur[t], acc[t]);
    a_cur = a_nxt;
#pragma unroll
    for (int t = 0; t < 4; ++t) b_cur[t] = b_nxt[t];
  }
}

// ---------------------------------------------------------------------------
// Utility kernels
// ---------------------------------------------------------------------------
__global__ void zero_h_kernel(_Float16* p, size_t n) {
  size_t i = (size_t)blockIdx.x * blockDim.x + threadIdx.x;
  size_t st = (size_t)gridDim.x * blockDim.x;
  for (; i < n; i += st) p[i] = (_Float16)0.f;
}

// fp32 [K][N] -> packed f16 fragments [Ktiles][Ntiles][32][16], zero pad.
__global__ void pack_weight_kernel(const float* __restrict__ src, int K, int N,
                                   _Float16* __restrict__ dst, int Ktiles,
                                   int Ntiles) {
  size_t tot = (size_t)Ktiles * Ntiles * 512;
  size_t i = (size_t)blockIdx.x * blockDim.x + threadIdx.x;
  size_t st = (size_t)gridDim.x * blockDim.x;
  for (; i < tot; i += st) {
    int j = (int)(i & 15);
    int lane = (int)((i >> 4) & 31);
    size_t blk = i >> 9;
    int nt = (int)(blk % Ntiles), kt = (int)(blk / Ntiles);
    int k = kt * 32 + ((lane >> 4) * 16) + j;
    int n = nt * 16 + (lane & 15);
    dst[i] = (k < K && n < N) ? (_Float16)src[(size_t)k * N + n] : (_Float16)0.f;
  }
}

// ---------------------------------------------------------------------------
// LN1 + roll(-1,-4,-4) + window partition.  One wave per window-token.
// ---------------------------------------------------------------------------
__global__ __launch_bounds__(256) void ln_win_kernel(
    const float* __restrict__ x, const float* __restrict__ g,
    const float* __restrict__ b, const float* __restrict__ pb,
    _Float16* __restrict__ XW, _Float16* __restrict__ XWPB) {
  int m = blockIdx.x * 8 + (threadIdx.x >> 5);
  int lane = threadIdx.x & 31;
  int w = m >> 7, tok = m & 127;
  int wd = w >> 8, wh = (w >> 4) & 15, ww = w & 15;
  int ld = tok >> 6, lh = (tok >> 3) & 7, lw = tok & 7;
  int d0 = (wd * 2 + ld + 1) % 6;          // roll -SS then read => +SS mod
  int h0 = (wh * 8 + lh + 4) & 127;
  int w0 = (ww * 8 + lw + 4) & 127;
  const float* src = x + (((size_t)d0 * 128 + h0) * 128 + w0) * CDIM;
  float s = 0.f, ss = 0.f;
  for (int c = lane; c < CDIM; c += 32) { float v = src[c]; s += v; ss += v * v; }
  for (int o = 16; o; o >>= 1) { s += __shfl_xor(s, o, 32); ss += __shfl_xor(ss, o, 32); }
  float mean = s * (1.f / CDIM);
  float inv = rsqrtf(ss * (1.f / CDIM) - mean * mean + 1e-5f);
  const float* pbr = pb + (size_t)(tok & 63) * CDIM;   // pb = concat(pos,pos)
  _Float16* o1 = XW + (size_t)m * CPAD;
  _Float16* o2 = XWPB + (size_t)m * CPAD;
  for (int c = lane; c < CDIM; c += 32) {
    float v = (src[c] - mean) * inv * g[c] + b[c];
    o1[c] = (_Float16)v;
    o2[c] = (_Float16)(v + pbr[c]);
  }
  if (lane < CPAD - CDIM) { o1[CDIM + lane] = (_Float16)0.f; o2[CDIM + lane] = (_Float16)0.f; }
}

// LN2 over residual y1, straight token order
__global__ __launch_bounds__(256) void ln2_kernel(
    const float* __restrict__ y1, const float* __restrict__ g,
    const float* __restrict__ b, _Float16* __restrict__ HH) {
  int m = blockIdx.x * 8 + (threadIdx.x >> 5);
  int lane = threadIdx.x & 31;
  const float* src = y1 + (size_t)m * CDIM;
  float s = 0.f, ss = 0.f;
  for (int c = lane; c < CDIM; c += 32) { float v = src[c]; s += v; ss += v * v; }
  for (int o = 16; o; o >>= 1) { s += __shfl_xor(s, o, 32); ss += __shfl_xor(ss, o, 32); }
  float mean = s * (1.f / CDIM);
  float inv = rsqrtf(ss * (1.f / CDIM) - mean * mean + 1e-5f);
  _Float16* o1 = HH + (size_t)m * CPAD;
  for (int c = lane; c < CDIM; c += 32)
    o1[c] = (_Float16)((src[c] - mean) * inv * g[c] + b[c]);
  if (lane < CPAD - CDIM) o1[CDIM + lane] = (_Float16)0.f;
}

// ---------------------------------------------------------------------------
// QKV GEMM: [MTOT][192] x packed[6][36] -> Q/K into [w][h][tok][32],
// V into packed-B fragment layout [w][h][kt(4)][nt(2)][32][16].
// ---------------------------------------------------------------------------
__global__ __launch_bounds__(128) void qkv_gemm_kernel(
    const _Float16* __restrict__ A, const _Float16* __restrict__ WP,
    const float* __restrict__ bias, _Float16* __restrict__ Qd,
    _Float16* __restrict__ Kd, _Float16* __restrict__ Vp) {
  int mt = blockIdx.x;
  int grp = blockIdx.y * blockDim.y + threadIdx.y;   // 0..8 (guarded)
  if (grp >= QKV_NT / 4) return;                     // wave-uniform
  int m0 = mt * 16;
  __builtin_prefetch(A + (size_t)(m0 + 16) * CPAD, 0, 1);
  v8f acc[4] = {{}, {}, {}, {}};
  gemm_strip<CKT, QKV_NT>(A, CPAD, m0, WP, grp * 4, acc);
  int lane = threadIdx.x & 31;
#pragma unroll
  for (int t = 0; t < 4; ++t) {
    int n = (grp * 4 + t) * 16 + (lane & 15);
    if (n < NQKV) {
      float bs = bias[n];
      int qi = n / CDIM, rem = n % CDIM, h = rem / HD, hd = rem % HD;
#pragma unroll
      for (int r = 0; r < 8; ++r) {
        int m = m0 + ((lane >> 4) * 8) + r;
        int w = m >> 7, tok = m & 127;
        _Float16 val = (_Float16)(acc[t][r] + bs);
        if (qi == 2) {   // V -> packed-B fragment layout
          int ktile = tok >> 5, lhi = (tok >> 4) & 1, j = tok & 15;
          int ntile = hd >> 4, dlane = lhi * 16 + (hd & 15);
          Vp[(((((size_t)w * NHEAD + h) * 4 + ktile) * 2 + ntile) * 32 + dlane) * 16 + j] = val;
        } else {
          _Float16* dst = (qi == 0) ? Qd : Kd;
          dst[(((size_t)w * NHEAD + h) * NTOK + tok) * HDP + hd] = val;
        }
      }
    }
  }
}

// ---------------------------------------------------------------------------
// Self attention: one block per (window, head); 4 waves.
// ---------------------------------------------------------------------------
__global__ __launch_bounds__(128) void self_attn_kernel(
    const _Float16* __restrict__ Q, const _Float16* __restrict__ K,
    const _Float16* __restrict__ Vp, const float* __restrict__ rpb,
    const int* __restrict__ rpi, const float* __restrict__ mask,
    _Float16* __restrict__ XO) {
  __shared__ float S[NTOK][NTOK];   // 64 KB
  int w = blockIdx.x, h = blockIdx.y;
  size_t base = ((size_t)w * NHEAD + h) * NTOK * HDP;
  const _Float16* q = Q + base;
  const _Float16* kk = K + base;
  const _Float16* vb = Vp + ((size_t)w * NHEAD + h) * 4096;  // 4kt*2nt*32*16
  int wave = threadIdx.x >> 5, lane = threadIdx.x & 31;

  // S = q k^T  (K dim = 32, one WMMA per 16x16 tile), one-ahead B loads
  for (int mi = 0; mi < 2; ++mi) {
    int mt = wave * 2 + mi;
    v16h a = load_A_frag(q, HDP, mt * 16, 0);
    v16h b_cur = load_BT_frag(kk, HDP, 0);
#pragma unroll
    for (int nt = 0; nt < 8; ++nt) {
      int ntn = (nt + 1 < 8) ? nt + 1 : nt;
      v16h b_nxt = load_BT_frag(kk, HDP, ntn * 16);
      v8f acc = {};
      acc = wmma16(a, b_cur, acc);
#pragma unroll
      for (int r = 0; r < 8; ++r)
        S[mt * 16 + ((lane >> 4) * 8) + r][nt * 16 + (lane & 15)] = acc[r];
      b_cur = b_nxt;
    }
  }
  __syncthreads();

  // scale + rel-pos bias + mask + row softmax (thread-per-row)
  {
    int i = threadIdx.x;
    const float* mrow = mask + ((size_t)w * NTOK + i) * NTOK;
    const int* rrow = rpi + i * NTOK;
    float mx = -3.0e38f;
    for (int j = 0; j < NTOK; ++j) {
      float val = S[i][j] * SCALE_F + rpb[rrow[j] * NHEAD + h] + mrow[j];
      S[i][j] = val;
      mx = fmaxf(mx, val);
    }
    float sum = 0.f;
    for (int j = 0; j < NTOK; ++j) { float e = expf(S[i][j] - mx); S[i][j] = e; sum += e; }
    float inv = 1.f / sum;
    for (int j = 0; j < NTOK; ++j) S[i][j] *= inv;
  }
  __syncthreads();

  // O = P V  (P f32->f16 in loader, V in packed-B layout), pipelined
  for (int mi = 0; mi < 2; ++mi) {
    int mt = wave * 2 + mi;
    for (int nt = 0; nt < 2; ++nt) {
      v8f acc = {};
      v16h a_cur = load_A_frag_f32(&S[0][0], NTOK, mt * 16, 0);
      v16h b_cur = load_B_packed(vb, 2, 0, nt);
#pragma unroll
      for (int kt = 0; kt < 4; ++kt) {
        int ktn = (kt + 1 < 4) ? kt + 1 : kt;
        v16h a_nxt = load_A_frag_f32(&S[0][0], NTOK, mt * 16, ktn * 32);
        v16h b_nxt = load_B_packed(vb, 2, ktn, nt);
        acc = wmma16(a_cur, b_cur, acc);
        a_cur = a_nxt; b_cur = b_nxt;
      }
      int hd = nt * 16 + (lane & 15);
      if (hd < HD) {
#pragma unroll
        for (int r = 0; r < 8; ++r) {
          int m = mt * 16 + ((lane >> 4) * 8) + r;
          XO[((size_t)w * NTOK + m) * HID2P + CDIM + h * HD + hd] = (_Float16)acc[r];
        }
      }
    }
  }
}

// ---------------------------------------------------------------------------
// Mutual attention: one block per (window, head, direction); 2 waves, 64x64 S
// p=0: x1 = attn(q[64:], k[:64], v[:64]) -> out rows 0..63
// p=1: x2 = attn(q[:64], k[64:], v[64:]) -> out rows 64..127
// ---------------------------------------------------------------------------
__global__ __launch_bounds__(64) void mut_attn_kernel(
    const _Float16* __restrict__ Q, const _Float16* __restrict__ K,
    const _Float16* __restrict__ Vp, const float* __restrict__ mask,
    _Float16* __restrict__ XO) {
  __shared__ float S[64][64];   // 16 KB
  int w = blockIdx.x, h = blockIdx.y, p = blockIdx.z;
  size_t base = ((size_t)w * NHEAD + h) * NTOK * HDP;
  const _Float16* q  = Q + base + (p == 0 ? 64 * HDP : 0);
  const _Float16* kk = K + base + (p == 0 ? 0 : 64 * HDP);
  const _Float16* vb = Vp + ((size_t)w * NHEAD + h) * 4096;
  int koff = (p == 0) ? 0 : 2;           // V k-tile offset into packed layout
  int outbase = (p == 0) ? 0 : 64;
  int wave = threadIdx.x >> 5, lane = threadIdx.x & 31;

  for (int mi = 0; mi < 2; ++mi) {
    int mt = wave * 2 + mi;
    v16h a = load_A_frag(q, HDP, mt * 16, 0);
    v16h b_cur = load_BT_frag(kk, HDP, 0);
#pragma unroll
    for (int nt = 0; nt < 4; ++nt) {
      int ntn = (nt + 1 < 4) ? nt + 1 : nt;
      v16h b_nxt = load_BT_frag(kk, HDP, ntn * 16);
      v8f acc = {};
      acc = wmma16(a, b_cur, acc);
#pragma unroll
      for (int r = 0; r < 8; ++r)
        S[mt * 16 + ((lane >> 4) * 8) + r][nt * 16 + (lane & 15)] = acc[r];
      b_cur = b_nxt;
    }
  }
  __syncthreads();
  {
    int i = threadIdx.x;   // 0..63
    const float* mrow = mask + ((size_t)w * NTOK + i) * NTOK;  // mask[:, :64, :64]
    float mx = -3.0e38f;
    for (int j = 0; j < 64; ++j) {
      float val = S[i][j] * SCALE_F + mrow[j];
      S[i][j] = val;
      mx = fmaxf(mx, val);
    }
    float sum = 0.f;
    for (int j = 0; j < 64; ++j) { float e = expf(S[i][j] - mx); S[i][j] = e; sum += e; }
    float inv = 1.f / sum;
    for (int j = 0; j < 64; ++j) S[i][j] *= inv;
  }
  __syncthreads();
  for (int mi = 0; mi < 2; ++mi) {
    int mt = wave * 2 + mi;
    for (int nt = 0; nt < 2; ++nt) {
      v8f acc = {};
      v16h a_cur = load_A_frag_f32(&S[0][0], 64, mt * 16, 0);
      v16h b_cur = load_B_packed(vb, 2, koff, nt);
#pragma unroll
      for (int kt = 0; kt < 2; ++kt) {
        int ktn = (kt + 1 < 2) ? kt + 1 : kt;
        v16h a_nxt = load_A_frag_f32(&S[0][0], 64, mt * 16, ktn * 32);
        v16h b_nxt = load_B_packed(vb, 2, koff + ktn, nt);
        acc = wmma16(a_cur, b_cur, acc);
        a_cur = a_nxt; b_cur = b_nxt;
      }
      int hd = nt * 16 + (lane & 15);
      if (hd < HD) {
#pragma unroll
        for (int r = 0; r < 8; ++r) {
          int m = mt * 16 + ((lane >> 4) * 8) + r;
          XO[((size_t)w * NTOK + outbase + m) * HID2P + h * HD + hd] = (_Float16)acc[r];
        }
      }
    }
  }
}

// ---------------------------------------------------------------------------
// Proj GEMM [MTOT][384] x packed[12][12] + bias, fused win_rev + roll + resid
// ---------------------------------------------------------------------------
__global__ __launch_bounds__(128) void proj_gemm_kernel(
    const _Float16* __restrict__ XO, const _Float16* __restrict__ WP,
    const float* __restrict__ bias, const float* __restrict__ x,
    float* __restrict__ y1) {
  int mt = blockIdx.x;
  int grp = threadIdx.y;                 // 0..3, need 0..2
  if (grp >= PRJ_NT / 4) return;         // wave-uniform
  int m0 = mt * 16;
  __builtin_prefetch(XO + (size_t)(m0 + 16) * HID2P, 0, 1);
  v8f acc[4] = {{}, {}, {}, {}};
  gemm_strip<H2KT, PRJ_NT>(XO, HID2P, m0, WP, grp * 4, acc);
  int lane = threadIdx.x & 31;
#pragma unroll
  for (int t = 0; t < 4; ++t) {
    int n = (grp * 4 + t) * 16 + (lane & 15);
    if (n < CDIM) {
      float bv = bias[n];
#pragma unroll
      for (int r = 0; r < 8; ++r) {
        int m = m0 + ((lane >> 4) * 8) + r;
        int w = m >> 7, tok = m & 127;
        int wd = w >> 8, wh = (w >> 4) & 15, ww = w & 15;
        int ld = tok >> 6, lh = (tok >> 3) & 7, lw = tok & 7;
        int d0 = (wd * 2 + ld + 1) % 6;
        int h0 = (wh * 8 + lh + 4) & 127;
        int w0 = (ww * 8 + lw + 4) & 127;
        size_t idx = (((size_t)d0 * 128 + h0) * 128 + w0) * CDIM + n;
        y1[idx] = x[idx] + acc[t][r] + bv;
      }
    }
  }
}

// ---------------------------------------------------------------------------
// MLP GEMM1 (dual): G = gelu(h@W11+b11) * (h@W12+b12) (f16)
// Pipelined 16x32 strips: 2 N-tiles x 2 weight mats = 4 WMMA per A fragment.
// ---------------------------------------------------------------------------
__global__ __launch_bounds__(128) void mlp1_gemm_kernel(
    const _Float16* __restrict__ A, const _Float16* __restrict__ WP1,
    const _Float16* __restrict__ WP2, const float* __restrict__ b11,
    const float* __restrict__ b12, _Float16* __restrict__ G) {
  int mt = blockIdx.x;
  int grp = blockIdx.y * blockDim.y + threadIdx.y;   // 0..11 exact
  int m0 = mt * 16;
  __builtin_prefetch(A + (size_t)(m0 + 16) * CPAD, 0, 1);
  v8f a1[2] = {{}, {}}, a2[2] = {{}, {}};
  v16h a_cur = load_A_frag(A, CPAD, m0, 0);
  v16h b1c[2], b2c[2];
#pragma unroll
  for (int t = 0; t < 2; ++t) {
    b1c[t] = load_B_packed(WP1, MLP_NT, 0, grp * 2 + t);
    b2c[t] = load_B_packed(WP2, MLP_NT, 0, grp * 2 + t);
  }
#pragma unroll
  for (int kt = 0; kt < CKT; ++kt) {
    int ktn = (kt + 1 < CKT) ? kt + 1 : kt;
    v16h a_nxt = load_A_frag(A, CPAD, m0, ktn * 32);
    v16h b1n[2], b2n[2];
#pragma unroll
    for (int t = 0; t < 2; ++t) {
      b1n[t] = load_B_packed(WP1, MLP_NT, ktn, grp * 2 + t);
      b2n[t] = load_B_packed(WP2, MLP_NT, ktn, grp * 2 + t);
    }
#pragma unroll
    for (int t = 0; t < 2; ++t) {
      a1[t] = wmma16(a_cur, b1c[t], a1[t]);
      a2[t] = wmma16(a_cur, b2c[t], a2[t]);
    }
    a_cur = a_nxt;
#pragma unroll
    for (int t = 0; t < 2; ++t) { b1c[t] = b1n[t]; b2c[t] = b2n[t]; }
  }
  int lane = threadIdx.x & 31;
#pragma unroll
  for (int t = 0; t < 2; ++t) {
    int n = (grp * 2 + t) * 16 + (lane & 15);
    if (n < HID) {
      float bb1 = b11[n], bb2 = b12[n];
#pragma unroll
      for (int r = 0; r < 8; ++r) {
        int m = m0 + ((lane >> 4) * 8) + r;
        float u1 = a1[t][r] + bb1;
        float u2 = a2[t][r] + bb2;
        float ge = 0.5f * u1 * (1.f + erff(u1 * 0.70710678118654752f));
        G[(size_t)m * HID2P + n] = (_Float16)(ge * u2);
      }
    }
  }
}

// ---------------------------------------------------------------------------
// MLP GEMM2: out = y1 + G@W2 + b2
// ---------------------------------------------------------------------------
__global__ __launch_bounds__(128) void mlp2_gemm_kernel(
    const _Float16* __restrict__ G, const _Float16* __restrict__ WP,
    const float* __restrict__ bias, const float* __restrict__ y1,
    float* __restrict__ out) {
  int mt = blockIdx.x;
  int grp = threadIdx.y;                 // 0..3, need 0..2
  if (grp >= PRJ_NT / 4) return;         // wave-uniform
  int m0 = mt * 16;
  __builtin_prefetch(G + (size_t)(m0 + 16) * HID2P, 0, 1);
  v8f acc[4] = {{}, {}, {}, {}};
  gemm_strip<H2KT, PRJ_NT>(G, HID2P, m0, WP, grp * 4, acc);
  int lane = threadIdx.x & 31;
#pragma unroll
  for (int t = 0; t < 4; ++t) {
    int n = (grp * 4 + t) * 16 + (lane & 15);
    if (n < CDIM) {
      float bv = bias[n];
#pragma unroll
      for (int r = 0; r < 8; ++r) {
        int m = m0 + ((lane >> 4) * 8) + r;
        size_t idx = (size_t)m * CDIM + n;
        out[idx] = y1[idx] + acc[t][r] + bv;
      }
    }
  }
}

// ---------------------------------------------------------------------------
extern "C" void kernel_launch(void* const* d_in, const int* in_sizes, int n_in,
                              void* d_out, int out_size, void* d_ws, size_t ws_size,
                              hipStream_t stream) {
  (void)in_sizes; (void)n_in; (void)out_size; (void)ws_size;
  const float* x          = (const float*)d_in[0];
  const float* mask       = (const float*)d_in[1];
  const float* g1         = (const float*)d_in[2];
  const float* b1         = (const float*)d_in[3];
  const float* g2         = (const float*)d_in[4];
  const float* b2         = (const float*)d_in[5];
  const float* w_qkv_self = (const float*)d_in[6];
  const float* b_qkv_self = (const float*)d_in[7];
  const float* w_qkv_mut  = (const float*)d_in[8];
  const float* b_qkv_mut  = (const float*)d_in[9];
  const float* rpb        = (const float*)d_in[10];
  const float* pos_bias   = (const float*)d_in[11];
  const float* w_proj     = (const float*)d_in[12];
  const float* b_proj     = (const float*)d_in[13];
  const float* w_fc11     = (const float*)d_in[14];
  const float* b_fc11     = (const float*)d_in[15];
  const float* w_fc12     = (const float*)d_in[16];
  const float* b_fc12     = (const float*)d_in[17];
  const float* w_fc2      = (const float*)d_in[18];
  const float* b_fc2      = (const float*)d_in[19];
  const int*   rpi        = (const int*)d_in[20];
  float* out = (float*)d_out;

  char* ws = (char*)d_ws;
  size_t off = 0;
  auto alloc = [&](size_t bytes) -> void* {
    void* p = (void*)(ws + off);
    off += (bytes + 255) & ~(size_t)255;
    return p;
  };

  const size_t QKVE = (size_t)NWIN * NHEAD * NTOK * HDP;   // per q/k/v array
  _Float16* xw_h   = (_Float16*)alloc((size_t)MTOT * CPAD * 2);
  _Float16* xwpb_h = (_Float16*)alloc((size_t)MTOT * CPAD * 2);
  _Float16* qs = (_Float16*)alloc(QKVE * 2);
  _Float16* ks = (_Float16*)alloc(QKVE * 2);
  _Float16* vs = (_Float16*)alloc(QKVE * 2);   // packed-B fragment layout
  _Float16* qm = (_Float16*)alloc(QKVE * 2);
  _Float16* km = (_Float16*)alloc(QKVE * 2);
  _Float16* vm = (_Float16*)alloc(QKVE * 2);   // packed-B fragment layout
  _Float16* xo_h = (_Float16*)alloc((size_t)MTOT * HID2P * 2);
  float*    y1   = (float*)alloc((size_t)MTOT * CDIM * 4);
  _Float16* hh_h = (_Float16*)alloc((size_t)MTOT * CPAD * 2);
  _Float16* g_h  = (_Float16*)alloc((size_t)MTOT * HID2P * 2);
  _Float16* Wqs = (_Float16*)alloc((size_t)CKT * QKV_NT * 512 * 2);
  _Float16* Wqm = (_Float16*)alloc((size_t)CKT * QKV_NT * 512 * 2);
  _Float16* Wp  = (_Float16*)alloc((size_t)H2KT * PRJ_NT * 512 * 2);
  _Float16* W11 = (_Float16*)alloc((size_t)CKT * MLP_NT * 512 * 2);
  _Float16* W12 = (_Float16*)alloc((size_t)CKT * MLP_NT * 512 * 2);
  _Float16* W2  = (_Float16*)alloc((size_t)H2KT * PRJ_NT * 512 * 2);

  // 1) weight pack (fragment layout) + pad
  pack_weight_kernel<<<512, 256, 0, stream>>>(w_qkv_self, CDIM, NQKV, Wqs, CKT, QKV_NT);
  pack_weight_kernel<<<512, 256, 0, stream>>>(w_qkv_mut,  CDIM, NQKV, Wqm, CKT, QKV_NT);
  pack_weight_kernel<<<512, 256, 0, stream>>>(w_proj, 2 * CDIM, CDIM, Wp, H2KT, PRJ_NT);
  pack_weight_kernel<<<512, 256, 0, stream>>>(w_fc11, CDIM, HID, W11, CKT, MLP_NT);
  pack_weight_kernel<<<512, 256, 0, stream>>>(w_fc12, CDIM, HID, W12, CKT, MLP_NT);
  pack_weight_kernel<<<512, 256, 0, stream>>>(w_fc2, HID, CDIM, W2, H2KT, PRJ_NT);

  // 2) zero padded activation buffers (hd pads + xo/g col pads)
  zero_h_kernel<<<2048, 256, 0, stream>>>(qs, QKVE);
  zero_h_kernel<<<2048, 256, 0, stream>>>(ks, QKVE);
  zero_h_kernel<<<2048, 256, 0, stream>>>(vs, QKVE);
  zero_h_kernel<<<2048, 256, 0, stream>>>(qm, QKVE);
  zero_h_kernel<<<2048, 256, 0, stream>>>(km, QKVE);
  zero_h_kernel<<<2048, 256, 0, stream>>>(vm, QKVE);
  zero_h_kernel<<<2048, 256, 0, stream>>>(xo_h, (size_t)MTOT * HID2P);
  zero_h_kernel<<<2048, 256, 0, stream>>>(g_h, (size_t)MTOT * HID2P);

  // 3) LN1 + roll + window partition (+pos_bias variant)
  ln_win_kernel<<<MTOT / 8, 256, 0, stream>>>(x, g1, b1, pos_bias, xw_h, xwpb_h);

  // 4) QKV GEMMs (WMMA, pipelined 16x64 strips)
  {
    dim3 grid(MTOT / 16, 3), blk(32, 4);   // 12 wave-slots, 9 groups guarded
    qkv_gemm_kernel<<<grid, blk, 0, stream>>>(xw_h, Wqs, b_qkv_self, qs, ks, vs);
    qkv_gemm_kernel<<<grid, blk, 0, stream>>>(xwpb_h, Wqm, b_qkv_mut, qm, km, vm);
  }

  // 5) attention (WMMA S=QK^T and O=PV)
  self_attn_kernel<<<dim3(NWIN, NHEAD), 128, 0, stream>>>(qs, ks, vs, rpb, rpi, mask, xo_h);
  mut_attn_kernel<<<dim3(NWIN, NHEAD, 2), 64, 0, stream>>>(qm, km, vm, mask, xo_h);

  // 6) proj + win_rev + un-roll + residual
  proj_gemm_kernel<<<dim3(MTOT / 16, 1), dim3(32, 4), 0, stream>>>(xo_h, Wp, b_proj, x, y1);

  // 7) LN2
  ln2_kernel<<<MTOT / 8, 256, 0, stream>>>(y1, g2, b2, hh_h);

  // 8) gated MLP
  mlp1_gemm_kernel<<<dim3(MTOT / 16, 3), dim3(32, 4), 0, stream>>>(hh_h, W11, W12,
                                                                   b_fc11, b_fc12, g_h);
  mlp2_gemm_kernel<<<dim3(MTOT / 16, 1), dim3(32, 4), 0, stream>>>(g_h, W2, b_fc2, y1, out);
}
